// WanAF2VCrossAttention_77232101916895
// MI455X (gfx1250) — compile-verified, matrix operand order checked by reading, not run on hardware
//
#include <hip/hip_runtime.h>

typedef __attribute__((ext_vector_type(16))) _Float16 v16h;
typedef __attribute__((ext_vector_type(8)))  float    v8f;

#define C_DIM   2048
#define H_NUM   16
#define D_DIM   128
#define L1_Q    8192
#define L2_TXT  512
#define LI_IMG  257
#define LA_AUD  512
#define LF_FACE 64
#define P_NUM   2

// Pack two f32 -> packed f16x2 dword (v_cvt_pk_rtz_f16_f32).
__device__ __forceinline__ unsigned pkh2(float a, float b) {
  auto p = __builtin_amdgcn_cvt_pkrtz(a, b);
  union { decltype(p) v; unsigned u; } t;
  t.v = p;
  return t.u;
}

// ---------------------------------------------------------------------------
// Fragment loader for 16-bit A/B WMMA operands (ISA 7.12.2 layout):
// lane L: row = L&15 ; element e -> k = ((e&8)<<1) + (e&7) + (L>>4)*8
// Pairs are contiguous in k; rows are 16B aligned so these merge to b128.
// ---------------------------------------------------------------------------
__device__ __forceinline__ v16h frag_load(const _Float16* rp, int hi) {
  rp += hi * 8;
  v16h a;
#pragma unroll
  for (int p = 0; p < 8; ++p) {
    int k = (p < 4) ? (p * 2) : (16 + (p - 4) * 2);
    a[2 * p]     = rp[k];
    a[2 * p + 1] = rp[k + 1];
  }
  return a;
}

// ---------------------------------------------------------------------------
// GEMM: Out[M,N] = A[M,K] @ W[K,N] + bias  (f32 in, f16 WMMA, f32 accum)
// Block 256 thr (8 waves), tile 128x128, BK=32; wave grid 4(M) x 2(N).
// ---------------------------------------------------------------------------
__global__ __launch_bounds__(256) void gemm_bias_f32(
    const float* __restrict__ A, const float* __restrict__ W,
    const float* __restrict__ bias, float* __restrict__ Out,
    int M, int N, int K)
{
  __shared__ _Float16 sA[128][40];   // [m][k]   row = 80B (16B aligned)
  __shared__ _Float16 sWt[128][40];  // [n][k]   (W transposed)

  const int tid  = threadIdx.x;
  const int lane = tid & 31, wave = tid >> 5;
  const int hi = lane >> 4, ln = lane & 15;
  const int wm = wave & 3,  wn = wave >> 2;
  const int m0 = blockIdx.x * 128, n0 = blockIdx.y * 128;

  v8f c[2][4];
#pragma unroll
  for (int i = 0; i < 2; ++i)
#pragma unroll
    for (int j = 0; j < 4; ++j)
#pragma unroll
      for (int r = 0; r < 8; ++r) c[i][j][r] = 0.f;

  for (int k0 = 0; k0 < K; k0 += 32) {
    { // ---- A tile 128x32: 16 floats/thread, single uniform bounds branch
      int row = tid >> 1;
      int col = (tid & 1) * 16;
      int gr  = m0 + row;
      uint4 u0, u1;
      if (gr < M) {
        const float4* ap = (const float4*)(A + (size_t)gr * K + k0 + col);
        float4 x0 = ap[0], x1 = ap[1], x2 = ap[2], x3 = ap[3];
        u0.x = pkh2(x0.x, x0.y); u0.y = pkh2(x0.z, x0.w);
        u0.z = pkh2(x1.x, x1.y); u0.w = pkh2(x1.z, x1.w);
        u1.x = pkh2(x2.x, x2.y); u1.y = pkh2(x2.z, x2.w);
        u1.z = pkh2(x3.x, x3.y); u1.w = pkh2(x3.z, x3.w);
      } else {
        u0.x = u0.y = u0.z = u0.w = 0u;
        u1 = u0;
      }
      *(uint4*)&sA[row][col]     = u0;
      *(uint4*)&sA[row][col + 8] = u1;
    }
    { // ---- W tile 32x128 transposed: k-pairs -> packed half2 dword stores
      int kp = tid >> 4;        // k-pair 0..15
      int c8 = (tid & 15) * 8;  // col group of 8
      const float4* w0 = (const float4*)(W + (size_t)(k0 + 2 * kp)     * N + n0 + c8);
      const float4* w1 = (const float4*)(W + (size_t)(k0 + 2 * kp + 1) * N + n0 + c8);
      float4 a0 = w0[0], a1 = w0[1];
      float4 b0 = w1[0], b1 = w1[1];
      float f0[8] = {a0.x, a0.y, a0.z, a0.w, a1.x, a1.y, a1.z, a1.w};
      float f1[8] = {b0.x, b0.y, b0.z, b0.w, b1.x, b1.y, b1.z, b1.w};
#pragma unroll
      for (int j = 0; j < 8; ++j)
        *(unsigned*)&sWt[c8 + j][2 * kp] = pkh2(f0[j], f1[j]);
    }
    __syncthreads();

    v16h a[2], b[4];
#pragma unroll
    for (int i = 0; i < 2; ++i) a[i] = frag_load(&sA[wm * 32 + i * 16 + ln][0], hi);
#pragma unroll
    for (int j = 0; j < 4; ++j) b[j] = frag_load(&sWt[wn * 64 + j * 16 + ln][0], hi);
#pragma unroll
    for (int i = 0; i < 2; ++i)
#pragma unroll
      for (int j = 0; j < 4; ++j)
        c[i][j] = __builtin_amdgcn_wmma_f32_16x16x32_f16(
            false, a[i], false, b[j], (short)0, c[i][j], false, false);
    __syncthreads();
  }

  // epilogue: C layout -> VGPR r = row (+hi*8), lane ln = col
#pragma unroll
  for (int i = 0; i < 2; ++i)
#pragma unroll
    for (int r = 0; r < 8; ++r) {
      int row = m0 + wm * 32 + i * 16 + hi * 8 + r;
      if (row < M) {
#pragma unroll
        for (int j = 0; j < 4; ++j) {
          int col = n0 + wn * 64 + j * 16 + ln;
          Out[(size_t)row * N + col] = c[i][j][r] + bias[col];
        }
      }
    }
}

// ---------------------------------------------------------------------------
// RMSNorm (fp32, eps=1e-6) * gain, cast to f16.  One block per C=2048 row.
// ---------------------------------------------------------------------------
__global__ __launch_bounds__(256) void rmsnorm_cast_f16(
    const float* __restrict__ Y, const float* __restrict__ g,
    _Float16* __restrict__ Out)
{
  __shared__ float wsum[8];
  __shared__ float tot;
  const int row = blockIdx.x, tid = threadIdx.x;
  const float4* y4 = (const float4*)(Y + (size_t)row * C_DIM + tid * 8);
  float4 v0 = y4[0], v1 = y4[1];
  float s = v0.x * v0.x + v0.y * v0.y + v0.z * v0.z + v0.w * v0.w +
            v1.x * v1.x + v1.y * v1.y + v1.z * v1.z + v1.w * v1.w;
#pragma unroll
  for (int off = 1; off < 32; off <<= 1) s += __shfl_xor(s, off, 32);
  if ((tid & 31) == 0) wsum[tid >> 5] = s;
  __syncthreads();
  if (tid == 0) {
    float t = 0.f;
    for (int w = 0; w < 8; ++w) t += wsum[w];
    tot = t;
  }
  __syncthreads();
  const float sc = rsqrtf(tot / (float)C_DIM + 1e-6f);
  const float4* g4 = (const float4*)(g + tid * 8);
  float4 g0 = g4[0], g1 = g4[1];
  uint4 u;
  u.x = pkh2(v0.x * sc * g0.x, v0.y * sc * g0.y);
  u.y = pkh2(v0.z * sc * g0.z, v0.w * sc * g0.w);
  u.z = pkh2(v1.x * sc * g1.x, v1.y * sc * g1.y);
  u.w = pkh2(v1.z * sc * g1.z, v1.w * sc * g1.w);
  *(uint4*)(Out + (size_t)row * C_DIM + tid * 8) = u;
}

// n must be a multiple of 4 (always a multiple of C_DIM here).
__global__ void cast_f32_f16(const float* __restrict__ in,
                             _Float16* __restrict__ out, int n)
{
  int i = (blockIdx.x * blockDim.x + threadIdx.x) * 4;
  if (i < n) {
    float4 f = *(const float4*)(in + i);
    uint2 u;
    u.x = pkh2(f.x, f.y);
    u.y = pkh2(f.z, f.w);
    *(uint2*)(out + i) = u;
  }
}

// ---------------------------------------------------------------------------
// Fused flash cross-attention. Block = 8 waves; wave w owns q rows
// [bx*128 + w*16, +16) of head by.  K tile is DMA'd into LDS with
// global_load_async_to_lds_b128 (ASYNCcnt); V tile is transposed through
// registers with packed half2 stores.  Online softmax per 16-lane half via
// shfl_xor; P re-laid-out through per-wave LDS (s_wait_dscnt) to become the
// A fragment of P@V.  Epilogue: Acc (=|+=) weight(row) * O / l.
// ---------------------------------------------------------------------------
__global__ __launch_bounds__(256) void attn_fused(
    const _Float16* __restrict__ Q,   // [L1, H*D] f16
    const _Float16* __restrict__ K,   // [Lk, H*D] f16 (RMS-normed)
    const _Float16* __restrict__ V,   // [Lk, H*D] f16
    float* __restrict__ Acc,          // [L1, H*D] f32
    int Lk,
    const int* __restrict__ lens,              // nullable, lens[0] clamps keys
    const unsigned char* __restrict__ tmask,   // nullable, keys >= toff gated
    int toff,
    const float* __restrict__ wrow,            // nullable per-query weight
    int accumulate)
{
  __shared__ _Float16 sK[32][136];     // [key][d]   row = 272B (16B aligned)
  __shared__ _Float16 sVt[128][40];    // [d][key]   row = 80B  (16B aligned)
  __shared__ _Float16 sP[8][16][40];   // per-wave P staging

  const int tid  = threadIdx.x;
  const int lane = tid & 31, wave = tid >> 5;
  const int hi = lane >> 4, ln = lane & 15;
  const int head = blockIdx.y;
  const int q0 = blockIdx.x * 128 + wave * 16;

  int vlen = Lk;
  if (lens) { int l = lens[0]; vlen = l < Lk ? l : Lk; }

  // Q fragments: 16 rows x 128 (D) = 4 A-frags
  const _Float16* Qh = Q + (size_t)head * D_DIM;
  v16h aq[4];
#pragma unroll
  for (int kk = 0; kk < 4; ++kk)
    aq[kk] = frag_load(Qh + (size_t)(q0 + ln) * C_DIM + kk * 32, hi);

  v8f o[8];
#pragma unroll
  for (int f = 0; f < 8; ++f)
#pragma unroll
    for (int r = 0; r < 8; ++r) o[f][r] = 0.f;
  float mrow[8], lrow[8];
#pragma unroll
  for (int r = 0; r < 8; ++r) { mrow[r] = -1e30f; lrow[r] = 0.f; }

  const float scale = 0.088388347648318447f; // 1/sqrt(128)

  for (int kb = 0; kb < Lk; kb += 32) {
    { // ---- K tile 32x128: async DMA straight into LDS (16B per lane x2)
      int r = tid >> 3;            // key 0..31
      int c = (tid & 7) * 16;      // d   0..112
      int kr = kb + r; if (kr >= Lk) kr = Lk - 1;   // clamp; masked below
      const _Float16* kp = K + (size_t)kr * C_DIM + head * D_DIM + c;
      unsigned lds0 = (unsigned)(uintptr_t)&sK[r][c];
      asm volatile("global_load_async_to_lds_b128 %0, %1, off"
                   :: "v"(lds0), "v"(kp) : "memory");
      asm volatile("global_load_async_to_lds_b128 %0, %1, off offset:16"
                   :: "v"(lds0), "v"(kp) : "memory");
    }
    { // ---- V tile transposed: key-pairs -> packed half2 dword stores
      int kp2 = tid >> 4;          // key pair 0..15
      int c8  = (tid & 15) * 8;    // d group of 8
      int kr0 = kb + 2 * kp2; if (kr0 >= Lk) kr0 = Lk - 1;
      int kr1 = kr0 + 1;      if (kr1 >= Lk) kr1 = Lk - 1;
      union { uint4 u; _Float16 h[8]; } a0, a1;
      a0.u = *(const uint4*)(V + (size_t)kr0 * C_DIM + head * D_DIM + c8);
      a1.u = *(const uint4*)(V + (size_t)kr1 * C_DIM + head * D_DIM + c8);
#pragma unroll
      for (int j = 0; j < 8; ++j) {
        union { _Float16 h[2]; unsigned u; } pr;
        pr.h[0] = a0.h[j];
        pr.h[1] = a1.h[j];
        *(unsigned*)&sVt[c8 + j][2 * kp2] = pr.u;
      }
    }
    // drain this wave's async DMA before the workgroup barrier
    asm volatile("s_wait_asynccnt 0" ::: "memory");
    __syncthreads();

    // ---- S = Q K^T for 32 keys (two 16-key C fragments)
    v8f s0, s1;
#pragma unroll
    for (int r = 0; r < 8; ++r) { s0[r] = 0.f; s1[r] = 0.f; }
#pragma unroll
    for (int kk = 0; kk < 4; ++kk) {
      v16h b0 = frag_load(&sK[ln][kk * 32], hi);
      v16h b1 = frag_load(&sK[16 + ln][kk * 32], hi);
      s0 = __builtin_amdgcn_wmma_f32_16x16x32_f16(false, aq[kk], false, b0, (short)0, s0, false, false);
      s1 = __builtin_amdgcn_wmma_f32_16x16x32_f16(false, aq[kk], false, b1, (short)0, s1, false, false);
    }

    // ---- masking + online softmax (lane = key column; VGPR r = query row)
    int key0 = kb + ln, key1 = kb + 16 + ln;
    bool v0 = key0 < vlen, v1 = key1 < vlen;
    if (tmask) {
      if (v0 && key0 >= toff) v0 = tmask[key0 - toff] != 0;
      if (v1 && key1 >= toff) v1 = tmask[key1 - toff] != 0;
    }
#pragma unroll
    for (int r = 0; r < 8; ++r) {
      float f0 = v0 ? s0[r] * scale : -1e30f;
      float f1 = v1 ? s1[r] * scale : -1e30f;
      float rm = fmaxf(f0, f1);
#pragma unroll
      for (int off = 1; off < 16; off <<= 1) rm = fmaxf(rm, __shfl_xor(rm, off, 32));
      float nm   = fmaxf(mrow[r], rm);
      float corr = __expf(mrow[r] - nm);
      float p0 = v0 ? __expf(f0 - nm) : 0.f;
      float p1 = v1 ? __expf(f1 - nm) : 0.f;
      float ps = p0 + p1;
#pragma unroll
      for (int off = 1; off < 16; off <<= 1) ps += __shfl_xor(ps, off, 32);
      lrow[r] = lrow[r] * corr + ps;
      mrow[r] = nm;
#pragma unroll
      for (int f = 0; f < 8; ++f) o[f][r] *= corr;
      int rr = hi * 8 + r;
      sP[wave][rr][ln]      = (_Float16)p0;
      sP[wave][rr][16 + ln] = (_Float16)p1;
    }
    // per-wave LDS write->read of sP (no cross-wave sharing; just drain DS)
    asm volatile("s_wait_dscnt 0" ::: "memory");

    // ---- O += P @ V  (A = P 16x32 keys, B = V 32keys x 16d per d-tile)
    v16h ap = frag_load(&sP[wave][ln][0], hi);
#pragma unroll
    for (int nt = 0; nt < 8; ++nt) {
      v16h bv = frag_load(&sVt[nt * 16 + ln][0], hi);
      o[nt] = __builtin_amdgcn_wmma_f32_16x16x32_f16(false, ap, false, bv, (short)0, o[nt], false, false);
    }
    __syncthreads();
  }

  // ---- epilogue
#pragma unroll
  for (int r = 0; r < 8; ++r) {
    int row = q0 + hi * 8 + r;
    float s = 1.f / fmaxf(lrow[r], 1e-20f);
    if (wrow) s *= wrow[row];
#pragma unroll
    for (int nt = 0; nt < 8; ++nt) {
      size_t idx = (size_t)row * C_DIM + head * D_DIM + nt * 16 + ln;
      float val = o[nt][r] * s;
      Acc[idx] = accumulate ? (Acc[idx] + val) : val;
    }
  }
}

// ---------------------------------------------------------------------------
extern "C" void kernel_launch(void* const* d_in, const int* in_sizes, int n_in,
                              void* d_out, int out_size, void* d_ws, size_t ws_size,
                              hipStream_t stream)
{
  (void)in_sizes; (void)n_in; (void)out_size; (void)ws_size;
  const float* x     = (const float*)d_in[0];
  const float* ctext = (const float*)d_in[1];
  const float* cimg  = (const float*)d_in[2];
  const float* caud  = (const float*)d_in[3];   // [P,B,LA,C]
  const float* cface = (const float*)d_in[4];   // [P,B,LF,C]
  const int*   lens  = (const int*)d_in[5];
  const unsigned char* tmask = (const unsigned char*)d_in[6];
  const float* fmask = (const float*)d_in[7];   // [P,B,L1,1]
  const float *Wq=(const float*)d_in[8],  *bq=(const float*)d_in[9];
  const float *Wk=(const float*)d_in[10], *bk=(const float*)d_in[11];
  const float *Wv=(const float*)d_in[12], *bv=(const float*)d_in[13];
  const float *Wo=(const float*)d_in[14], *bo=(const float*)d_in[15];
  const float *Wki=(const float*)d_in[16],*bki=(const float*)d_in[17];
  const float *Wvi=(const float*)d_in[18],*bvi=(const float*)d_in[19];
  const float *Wka=(const float*)d_in[20],*bka=(const float*)d_in[21];
  const float *Wva=(const float*)d_in[22],*bva=(const float*)d_in[23];
  const float *Wkf=(const float*)d_in[24],*bkf=(const float*)d_in[25];
  const float *Wvf=(const float*)d_in[26],*bvf=(const float*)d_in[27];
  const float *gq=(const float*)d_in[28], *gk=(const float*)d_in[29];
  const float *gki=(const float*)d_in[30],*gka=(const float*)d_in[31];
  const float *gkf=(const float*)d_in[32];

  char* ws = (char*)d_ws;
  size_t off = 0;
  auto wsalloc = [&](size_t bytes) -> void* {
    void* p = ws + off;
    off = (off + bytes + 255) & ~(size_t)255;
    return p;
  };
  float*    proj = (float*)   wsalloc((size_t)L1_Q * C_DIM * 4);  // GEMM scratch
  _Float16* qf   = (_Float16*)wsalloc((size_t)L1_Q * C_DIM * 2);
  float*    acc  = (float*)   wsalloc((size_t)L1_Q * C_DIM * 4);
  _Float16* ktxt = (_Float16*)wsalloc((size_t)L2_TXT * C_DIM * 2);
  _Float16* vtxt = (_Float16*)wsalloc((size_t)L2_TXT * C_DIM * 2);
  _Float16* kimg = (_Float16*)wsalloc((size_t)LI_IMG * C_DIM * 2);
  _Float16* vimg = (_Float16*)wsalloc((size_t)LI_IMG * C_DIM * 2);
  const int LKCAT = LF_FACE + LA_AUD;  // 576
  _Float16* kcat[P_NUM]; _Float16* vcat[P_NUM];
  for (int p = 0; p < P_NUM; ++p) {
    kcat[p] = (_Float16*)wsalloc((size_t)LKCAT * C_DIM * 2);
    vcat[p] = (_Float16*)wsalloc((size_t)LKCAT * C_DIM * 2);
  }

  dim3 blk(256);
  auto gemm = [&](const float* A, const float* W, const float* b, float* o, int M) {
    dim3 grd((M + 127) / 128, C_DIM / 128);
    gemm_bias_f32<<<grd, blk, 0, stream>>>(A, W, b, o, M, C_DIM, C_DIM);
  };
  auto rms = [&](const float* y, const float* g, _Float16* o, int rows) {
    rmsnorm_cast_f16<<<dim3(rows), blk, 0, stream>>>(y, g, o);
  };
  auto cast = [&](const float* in, _Float16* o, int n) {
    cast_f32_f16<<<dim3((n / 4 + 255) / 256), blk, 0, stream>>>(in, o, n);
  };

  // ---- projections -------------------------------------------------------
  gemm(x, Wq, bq, proj, L1_Q);          rms(proj, gq, qf, L1_Q);
  gemm(ctext, Wk, bk, proj, L2_TXT);    rms(proj, gk, ktxt, L2_TXT);
  gemm(ctext, Wv, bv, proj, L2_TXT);    cast(proj, vtxt, L2_TXT * C_DIM);
  gemm(cimg, Wki, bki, proj, LI_IMG);   rms(proj, gki, kimg, LI_IMG);
  gemm(cimg, Wvi, bvi, proj, LI_IMG);   cast(proj, vimg, LI_IMG * C_DIM);
  for (int p = 0; p < P_NUM; ++p) {
    const float* cf = cface + (size_t)p * LF_FACE * C_DIM;
    const float* ca = caud  + (size_t)p * LA_AUD  * C_DIM;
    gemm(cf, Wkf, bkf, proj, LF_FACE);  rms(proj, gkf, kcat[p], LF_FACE);
    gemm(cf, Wvf, bvf, proj, LF_FACE);  cast(proj, vcat[p], LF_FACE * C_DIM);
    gemm(ca, Wka, bka, proj, LA_AUD);   rms(proj, gka, kcat[p] + (size_t)LF_FACE * C_DIM, LA_AUD);
    gemm(ca, Wva, bva, proj, LA_AUD);   cast(proj, vcat[p] + (size_t)LF_FACE * C_DIM, LA_AUD * C_DIM);
  }

  // ---- attentions (sequential accumulation into acc) ---------------------
  dim3 agrd(L1_Q / 128, H_NUM);
  attn_fused<<<agrd, blk, 0, stream>>>(qf, ktxt, vtxt, acc, L2_TXT,
                                       lens, nullptr, 0, nullptr, /*acc=*/0);
  attn_fused<<<agrd, blk, 0, stream>>>(qf, kimg, vimg, acc, LI_IMG,
                                       nullptr, nullptr, 0, nullptr, 1);
  for (int p = 0; p < P_NUM; ++p)
    attn_fused<<<agrd, blk, 0, stream>>>(qf, kcat[p], vcat[p], acc, LKCAT,
                                         nullptr, tmask, LF_FACE,
                                         fmask + (size_t)p * L1_Q, 1);

  // ---- output projection -------------------------------------------------
  gemm(acc, Wo, bo, (float*)d_out, L1_Q);
}
